// GCNWorker_34892314312745
// MI455X (gfx1250) — matrix operand order repaired and checked by reading, not compiled, not run on hardware
//
#include <hip/hip_runtime.h>
#include <math.h>

#define NN 100000
#define NE 1600000
#define DF 128

typedef __attribute__((ext_vector_type(16))) __bf16 v16bf;
typedef __attribute__((ext_vector_type(8)))  __bf16 v8bf;
typedef __attribute__((ext_vector_type(8)))  float  v8f;

// ---------- normalization ----------
__global__ void sg_deg_init(float* deg) {
    int i = blockIdx.x * blockDim.x + threadIdx.x;
    if (i < NN) deg[i] = 1.0f;  // self-loop contribution
}

__global__ void sg_deg_count(const int* __restrict__ dst, float* deg) {
    int e = blockIdx.x * blockDim.x + threadIdx.x;
    if (e < NE) atomicAdd(&deg[dst[e]], 1.0f);
}

__global__ void sg_dinv(float* deg) {
    int i = blockIdx.x * blockDim.x + threadIdx.x;
    if (i < NN) {
        float d = deg[i];
        deg[i] = (d > 0.0f) ? rsqrtf(d) : 0.0f;
    }
}

// ---------- buffer zeroing (float4 granularity) ----------
__global__ void sg_zero4(float4* p, int n4) {
    int i = blockIdx.x * blockDim.x + threadIdx.x;
    if (i < n4) p[i] = make_float4(0.f, 0.f, 0.f, 0.f);
}

// ---------- edge scatter: one wave (32 lanes x float4) per edge ----------
__global__ void sg_scatter(const float4* __restrict__ h,
                           const int* __restrict__ src,
                           const int* __restrict__ dst,
                           const float* __restrict__ dinv,
                           float* __restrict__ out) {
    long long tid = (long long)blockIdx.x * blockDim.x + threadIdx.x;
    int e    = (int)(tid >> 5);
    int lane = (int)(tid & 31);
    if (e >= NE) return;
    int s = src[e];
    int d = dst[e];
    float nrm = dinv[s] * dinv[d];
    float4 v = h[(size_t)s * 32 + lane];
    float* o = out + (size_t)d * DF + lane * 4;
    atomicAdd(o + 0, v.x * nrm);
    atomicAdd(o + 1, v.y * nrm);
    atomicAdd(o + 2, v.z * nrm);
    atomicAdd(o + 3, v.w * nrm);
}

// ---------- self-loop term: out[i] += h[i] * dinv[i]^2 ----------
__global__ void sg_selfloop(const float4* __restrict__ h,
                            const float* __restrict__ dinv,
                            float4* __restrict__ out) {
    int i = blockIdx.x * blockDim.x + threadIdx.x;   // over NN*32 float4 chunks
    if (i >= NN * 32) return;
    int node = i >> 5;
    float w = dinv[node];
    w = w * w;
    float4 v = h[i];
    float4 o = out[i];
    o.x += v.x * w; o.y += v.y * w; o.z += v.z * w; o.w += v.w * w;
    out[i] = o;
}

// ---------- in-place WMMA GEMM: hout = hout @ W + b ----------
// Block = 256 threads = 8 waves; block b owns output rows [16b, 16b+16).
// W staged TRANSPOSED (column-major) as bf16 in LDS so each lane's B fragment
// is two 16-byte runs (k..k+7, k+16..k+23) -> two ds_load_b128.
#define WT_STRIDE 136   // pad: 272 B/row = 68 dwords -> 4-bank step per lane
__global__ __launch_bounds__(256)
void sg_gemm_wmma(float* __restrict__ hout,
                  const float* __restrict__ Wg,
                  const float* __restrict__ bias) {
    __shared__ __bf16 Wt[128 * WT_STRIDE];   // Wt[n][k] = W[k][n], bf16

    int tid = threadIdx.x;
    for (int i = tid; i < 128 * 128; i += 256) {
        int k = i >> 7;
        int n = i & 127;
        Wt[n * WT_STRIDE + k] = (__bf16)Wg[i];
    }

    int lane = tid & 31;
    int wv   = tid >> 5;          // wave id 0..7 -> N-tile
    int r    = lane & 15;
    int hi   = lane >> 4;         // lane half selects K sub-range

    size_t m = (size_t)blockIdx.x * 16 + r;
    const float4* hrow4 = (const float4*)(hout + m * DF);

    // A fragments (16-bit A 16x32 layout):
    // elems 0..7  -> K = s*32 + hi*8 + (0..7)
    // elems 8..15 -> K = s*32 + hi*8 + 16 + (0..7)
    v16bf afrag[4];
#pragma unroll
    for (int s = 0; s < 4; ++s) {
        int b4 = s * 8 + hi * 2;              // (s*32 + hi*8)/4
        float4 f0 = hrow4[b4];
        float4 f1 = hrow4[b4 + 1];
        float4 f2 = hrow4[b4 + 4];            // +16 floats
        float4 f3 = hrow4[b4 + 5];
        afrag[s][0]  = (__bf16)f0.x; afrag[s][1]  = (__bf16)f0.y;
        afrag[s][2]  = (__bf16)f0.z; afrag[s][3]  = (__bf16)f0.w;
        afrag[s][4]  = (__bf16)f1.x; afrag[s][5]  = (__bf16)f1.y;
        afrag[s][6]  = (__bf16)f1.z; afrag[s][7]  = (__bf16)f1.w;
        afrag[s][8]  = (__bf16)f2.x; afrag[s][9]  = (__bf16)f2.y;
        afrag[s][10] = (__bf16)f2.z; afrag[s][11] = (__bf16)f2.w;
        afrag[s][12] = (__bf16)f3.x; afrag[s][13] = (__bf16)f3.y;
        afrag[s][14] = (__bf16)f3.z; afrag[s][15] = (__bf16)f3.w;
    }

    // All global reads of hout must land before any thread overwrites it.
    asm volatile("s_wait_loadcnt 0" ::: "memory");
    __syncthreads();

    int n = wv * 16 + r;
    const __bf16* wcol = Wt + (size_t)n * WT_STRIDE;
    v8f c = {};
#pragma unroll
    for (int s = 0; s < 4; ++s) {
        // B fragment (32x16): lane holds col n, K = s*32 + hi*8 + (0..7), then s*32 + hi*8 + 16 + (0..7)
        int kb = s * 32 + hi * 8;
        v8bf blo = *(const v8bf*)(wcol + kb);        // ds_load_b128
        v8bf bhi = *(const v8bf*)(wcol + kb + 16);   // ds_load_b128
        v16bf bfrag = __builtin_shufflevector(blo, bhi,
            0, 1, 2, 3, 4, 5, 6, 7, 8, 9, 10, 11, 12, 13, 14, 15);
        c = __builtin_amdgcn_wmma_f32_16x16x32_bf16(
                false, afrag[s], false, bfrag, (short)0, c, false, false);
    }

    float bv = bias[n];
    size_t mb = (size_t)blockIdx.x * 16;
    // D layout: VGPR rr -> row (rr + 8*hi), col = n
#pragma unroll
    for (int rr = 0; rr < 8; ++rr) {
        hout[(mb + rr + hi * 8) * DF + n] = c[rr] + bv;
    }
}

extern "C" void kernel_launch(void* const* d_in, const int* in_sizes, int n_in,
                              void* d_out, int out_size, void* d_ws, size_t ws_size,
                              hipStream_t stream) {
    const float* x    = (const float*)d_in[0];
    const int*   eidx = (const int*)d_in[1];     // flat (2, NE): row0=src, row1=dst
    const float* W    = (const float*)d_in[2];
    const float* b    = (const float*)d_in[3];
    float*       out  = (float*)d_out;

    const int* src = eidx;
    const int* dst = eidx + NE;

    // workspace layout: dinv (NN floats, padded) | h1 (NN*DF floats)
    char* ws = (char*)d_ws;
    float* dinv = (float*)ws;
    size_t dinv_bytes = ((size_t)NN * 4 + 255) & ~(size_t)255;
    float* h1 = (float*)(ws + dinv_bytes);

    const int B = 256;
    int gN   = (NN + B - 1) / B;                 // node-parallel
    int gE   = (NE + B - 1) / B;                 // edge-parallel
    int gF4  = (NN * 32 + B - 1) / B;            // NN*DF/4 float4 chunks
    int gSc  = (int)(((long long)NE * 32 + B - 1) / B);  // wave-per-edge
    int gZ   = (NN * DF / 4 + B - 1) / B;

    // 1) normalization
    sg_deg_init<<<gN, B, 0, stream>>>(dinv);
    sg_deg_count<<<gE, B, 0, stream>>>(dst, dinv);
    sg_dinv<<<gN, B, 0, stream>>>(dinv);

    // 2) hop 1: h1 = scatter(x)
    sg_zero4<<<gZ, B, 0, stream>>>((float4*)h1, NN * DF / 4);
    sg_scatter<<<gSc, B, 0, stream>>>((const float4*)x, src, dst, dinv, h1);
    sg_selfloop<<<gF4, B, 0, stream>>>((const float4*)x, dinv, (float4*)h1);

    // 3) hop 2: out = scatter(h1)
    sg_zero4<<<gZ, B, 0, stream>>>((float4*)out, NN * DF / 4);
    sg_scatter<<<gSc, B, 0, stream>>>((const float4*)h1, src, dst, dinv, out);
    sg_selfloop<<<gF4, B, 0, stream>>>((const float4*)h1, dinv, (float4*)out);

    // 4) projection in place: out = out @ W + b   (100000 = 6250 * 16)
    sg_gemm_wmma<<<NN / 16, B, 0, stream>>>(out, W, b);
}
